// FullAttention_23270132809999
// MI455X (gfx1250) — matrix operand order repaired and checked by reading, not compiled, not run on hardware
//
#include <hip/hip_runtime.h>
#include <math.h>

// ---------------------------------------------------------------------------
// Flash-attention forward (causal), fp32 in/out, f16 WMMA math w/ fp32 accum.
// Shapes fixed by the reference: B=2, L=S=2048, H=8, E=D=64.
// Layout of q/k/v/out: (B, L, H, E) -> elem (b,l,h,e) at ((b*L+l)*H+h)*E+e.
//
// Per workgroup: 4 waves x 16 query rows = 64 rows; key blocks of 32 so the
// WMMA K-dim (32) is matched exactly.  Row sums of the probability tile are
// computed with an extra WMMA against a ones matrix (lands in D-layout right
// where the running softmax denominator lives); row maxima use ds_swizzle
// xor-butterflies (no bpermute address setup).
// ---------------------------------------------------------------------------

typedef __attribute__((ext_vector_type(16))) _Float16 v16h;
typedef __attribute__((ext_vector_type(8)))  float    v8f;

#define B_    2
#define L_    2048
#define H_    8
#define E_    64
#define SCALE 0.125f      // 1/sqrt(64)

// LDS row strides in f16 units; *2 bytes is a multiple of 16 so v16h
// (2 x ds_load_b128) accesses stay aligned, and strides break the
// power-of-two bank pattern.
#define KPAD 80           // K tile: 32 keys x 64 dims, row-major [key][e]
#define VPAD 40           // V tile transposed: 64 dims x 32 keys [d][key]
#define PPAD 40           // P tile per wave: 16 rows x 32 keys

// ds_swizzle group-of-32 encoding: and=0x1F, or=0, xor=m  -> lanes ^ m.
// masks <= 8 keep the butterfly inside each 16-lane half.
#define XOR_SWZ(m) (((m) << 10) | 0x1F)

__device__ __forceinline__ float hmax16(float v) {
  v = fmaxf(v, __int_as_float(__builtin_amdgcn_ds_swizzle(__float_as_int(v), XOR_SWZ(8))));
  v = fmaxf(v, __int_as_float(__builtin_amdgcn_ds_swizzle(__float_as_int(v), XOR_SWZ(4))));
  v = fmaxf(v, __int_as_float(__builtin_amdgcn_ds_swizzle(__float_as_int(v), XOR_SWZ(2))));
  v = fmaxf(v, __int_as_float(__builtin_amdgcn_ds_swizzle(__float_as_int(v), XOR_SWZ(1))));
  return v;
}

__global__ __launch_bounds__(128, 2)
void fa_fwd_causal(const float* __restrict__ Qp, const float* __restrict__ Kp,
                   const float* __restrict__ Vp, float* __restrict__ Op)
{
  __shared__ _Float16 sK [32 * KPAD];        // 5120 B
  __shared__ _Float16 sVt[64 * VPAD];        // 5120 B
  __shared__ _Float16 sP [4 * 16 * PPAD];    // 5120 B (wave-private slices)

  const int tid  = threadIdx.x;
  const int wave = tid >> 5;
  const int lane = tid & 31;
  const int n16  = lane & 15;                // N (col) / M (row) index in a half
  const int g    = lane >> 4;                // half-wave selector

  const int bh = blockIdx.y;
  const int b  = bh / H_;
  const int h  = bh % H_;

  const int qblk  = blockIdx.x;              // 64-row query block
  const int qbase = qblk * 64 + wave * 16;   // this wave's first query row
  // diagonal 32-key block index; uniform per wave via SGPR so the branches
  // around the WMMA region stay scalar (EXEC remains all-ones).
  const int kbD = __builtin_amdgcn_readfirstlane(qbase >> 5);
  const int nkb = 2 * qblk + 2;              // uniform loop bound for the WG

  const size_t rowStride = (size_t)H_ * E_;  // floats between seq positions
  const size_t base      = ((size_t)b * L_) * rowStride + (size_t)h * E_;

  // ---- Q tile -> registers in WMMA 16-bit A layout, pre-scaled ------------
  // A(16x32) order used here (contiguous k, same as B): lane = m + 16g; halfs 0..7 = K=16g..16g+7,
  //                                  halfs 8..15 = K=16g+8..16g+15.
  v16h qa[2];                                // es = E-slice (e = 32*es + k)
  {
    const float* qrow = Qp + base + (size_t)(qbase + n16) * rowStride;
#pragma unroll
    for (int es = 0; es < 2; ++es) {
      const float* c1 = qrow + es * 32 + 16 * g;
      const float* c2 = qrow + es * 32 + 16 * g + 8;
      v16h a;
#pragma unroll
      for (int i = 0; i < 8; ++i) {
        a[i]     = (_Float16)(c1[i] * SCALE);
        a[8 + i] = (_Float16)(c2[i] * SCALE);
      }
      qa[es] = a;
    }
  }

  // all-ones B matrix for the row-sum WMMA
  v16h ones;
#pragma unroll
  for (int i = 0; i < 16; ++i) ones[i] = (_Float16)1.0f;

  // ---- running softmax state and output accumulators ----------------------
  // C/D(16x16 f32) layout: VGPR r holds row r (lanes 0-15) / row r+8 (16-31),
  // lane n16 = column.  m_r[r] is that row's max, replicated per lane; lacc
  // is the running denominator in the same D layout (every lane of a row
  // carries the row sum after the ones-WMMA).
  v8f acc[4], lacc;
  float m_r[8];
#pragma unroll
  for (int j = 0; j < 4; ++j) acc[j] = (v8f)0.0f;
  lacc = (v8f)0.0f;
#pragma unroll
  for (int r = 0; r < 8; ++r) m_r[r] = -__builtin_inff();

  _Float16* myP = sP + wave * 16 * PPAD;

  // cooperative-loader coordinates: 128 threads cover 32 rows x 64 cols
  const int lr = tid >> 2;                   // key row 0..31
  const int lc = (tid & 3) * 16;             // col start 0,16,32,48

  // one key block of 32: 4 QK^T WMMAs + 1 row-sum WMMA + 4 PV WMMAs.
  auto process_block = [&](int kb, bool DIAG) {
    // S = (Q*scale) @ K^T : two 16x16 N-halves, accumulate over E.
    // B(32x16) order used here (contiguous k, same as A): lane = n + 16g; halfs 0..15 = K-rows 16g..16g+15.
    v8f S0 = (v8f)0.0f, S1 = (v8f)0.0f;
#pragma unroll
    for (int es = 0; es < 2; ++es) {
      v16h b0 = *(const v16h*)&sK[(0  + n16) * KPAD + es * 32 + 16 * g];
      v16h b1 = *(const v16h*)&sK[(16 + n16) * KPAD + es * 32 + 16 * g];
      S0 = __builtin_amdgcn_wmma_f32_16x16x32_f16(false, qa[es], false, b0,
                                                  (short)0, S0, false, false);
      S1 = __builtin_amdgcn_wmma_f32_16x16x32_f16(false, qa[es], false, b1,
                                                  (short)0, S1, false, false);
    }

    // online softmax over the 32 fresh columns
    const int c0 = kb * 32 + n16;
    const int c1 = c0 + 16;
#pragma unroll
    for (int r = 0; r < 8; ++r) {
      float s0 = S0[r], s1 = S1[r];
      if (DIAG) {                            // causal mask, diagonal block only
        const int row = qbase + r + 8 * g;
        s0 = (c0 > row) ? -__builtin_inff() : s0;
        s1 = (c1 > row) ? -__builtin_inff() : s1;
      }
      const float rmax = hmax16(fmaxf(s0, s1));
      const float mnew = fmaxf(m_r[r], rmax);
      const float corr = __expf(m_r[r] - mnew);
      m_r[r] = mnew;
      const float p0 = __expf(s0 - mnew);
      const float p1 = __expf(s1 - mnew);
      lacc[r] *= corr;
#pragma unroll
      for (int j = 0; j < 4; ++j) acc[j][r] *= corr;
      // D-layout -> row-major P tile in wave-private LDS (A-layout source)
      myP[(r + 8 * g) * PPAD + n16]      = (_Float16)p0;
      myP[(r + 8 * g) * PPAD + 16 + n16] = (_Float16)p1;
    }

    // reload P in A layout (compiler inserts s_wait_dscnt for the RAW dep)
    v16h aP;
    {
      const _Float16* pr = &myP[n16 * PPAD];
#pragma unroll
      for (int i = 0; i < 8; ++i) {
        aP[i]     = pr[16 * g + i];
        aP[8 + i] = pr[16 * g + 8 + i];
      }
    }
    // row sums: P @ ones -> every lane of row r gets sum(P[r][:]) in lacc[r]
    lacc = __builtin_amdgcn_wmma_f32_16x16x32_f16(false, aP, false, ones,
                                                  (short)0, lacc, false, false);
    // O += P @ V
#pragma unroll
    for (int j = 0; j < 4; ++j) {
      v16h bv = *(const v16h*)&sVt[(j * 16 + n16) * VPAD + 16 * g];
      acc[j] = __builtin_amdgcn_wmma_f32_16x16x32_f16(false, aP, false, bv,
                                                      (short)0, acc[j],
                                                      false, false);
    }
  };

  union H2 { _Float16 h[2]; unsigned u; };

  for (int kb = 0; kb < nkb; ++kb) {
    // ---- stage K (row-major) and V (transposed) tiles into LDS as f16 ----
    {
      const float* kg = Kp + base + (size_t)(kb * 32 + lr) * rowStride + lc;
      const float* vg = Vp + base + (size_t)(kb * 32 + lr) * rowStride + lc;
      unsigned* dstK = (unsigned*)&sK[lr * KPAD + lc];   // 4B-aligned (KPAD,lc even)
#pragma unroll
      for (int i = 0; i < 8; ++i) {
        H2 t; t.h[0] = (_Float16)kg[2 * i]; t.h[1] = (_Float16)kg[2 * i + 1];
        dstK[i] = t.u;
      }
#pragma unroll
      for (int i = 0; i < 16; ++i) sVt[(lc + i) * VPAD + lr] = (_Float16)vg[i];
      if (kb + 1 < nkb) {                    // gfx1250 global_prefetch_b8
        __builtin_prefetch(kg + 32 * rowStride, 0, 1);
        __builtin_prefetch(vg + 32 * rowStride, 0, 1);
      }
    }
    __syncthreads();

    if (kb < kbD)       process_block(kb, false);   // interior: no masking
    else if (kb == kbD) process_block(kb, true);    // diagonal: causal mask
    __syncthreads();
  }

  // ---- normalize and store O (fp32) ---------------------------------------
#pragma unroll
  for (int r = 0; r < 8; ++r) {
    const int row = qbase + r + 8 * g;
    const float inv = 1.0f / lacc[r];
    float* orow = Op + base + (size_t)row * rowStride;
#pragma unroll
    for (int j = 0; j < 4; ++j) orow[j * 16 + n16] = acc[j][r] * inv;
  }
}

extern "C" void kernel_launch(void* const* d_in, const int* in_sizes, int n_in,
                              void* d_out, int out_size, void* d_ws, size_t ws_size,
                              hipStream_t stream) {
  (void)in_sizes; (void)n_in; (void)out_size; (void)d_ws; (void)ws_size;
  const float* Q = (const float*)d_in[0];
  const float* K = (const float*)d_in[1];
  const float* V = (const float*)d_in[2];
  float*       O = (float*)d_out;
  dim3 grid(L_ / 64, B_ * H_);   // (32, 16): 512 workgroups
  fa_fwd_causal<<<grid, dim3(128), 0, stream>>>(Q, K, V, O);
}
